// KernelizedHeadAttention_29051158790769
// MI455X (gfx1250) — compile-verified, matrix operand order checked by reading, not run on hardware
//
#include <hip/hip_runtime.h>
#include <hip/hip_bf16.h>

// Problem constants (from the reference)
#define Bc    2
#define Sc    2048
#define Hc    16
#define DHc   64
#define DHIDc 256
#define DKERc 64
#define Dc    1024   // H*DH

typedef __bf16 bf16;
typedef __attribute__((ext_vector_type(16))) __bf16 v16bf;
typedef __attribute__((ext_vector_type(8)))  __bf16 v8bf;
typedef __attribute__((ext_vector_type(8)))  float  v8f;

__device__ __forceinline__ float gelu_tanh(float x) {
    // jax.nn.gelu(approximate=True)
    const float c = 0.7978845608028654f;
    float x3 = x * x * x;
    return 0.5f * x * (1.0f + tanhf(c * (x + 0.044715f * x3)));
}

// Load one bf16 A/B fragment half-row: 2 x 16B chunks per lane.
// rowptr points at the start of this lane's 32-wide K chunk of its row.
__device__ __forceinline__ v16bf load_frag(const bf16* rowptr, int koff) {
    union { v16bf v; v8bf h[2]; } u;
    u.h[0] = *(const v8bf*)(rowptr + koff);        // K = koff..koff+7
    u.h[1] = *(const v8bf*)(rowptr + 16 + koff);   // K = 16+koff..16+koff+7
    return u.v;
}

__device__ __forceinline__ v8f wmma_bf16(v16bf a, v16bf b, v8f c) {
    return __builtin_amdgcn_wmma_f32_16x16x32_bf16(
        false, a, false, b, (short)0, c, false, false);
}

// ---------------------------------------------------------------------------
// Kernel 0: bit-pack the boolean mask  [B,1,S,S] u8  ->  [B,S,S/32] u32.
// 1 MB result stays L2-resident; attention reads 1 u32 per 32-key tile
// instead of 16-24 byte loads.
// ---------------------------------------------------------------------------
__global__ __launch_bounds__(256) void maskpack_kernel(
    const unsigned char* __restrict__ mask, unsigned* __restrict__ mbits)
{
    // one output word (32 mask bytes) per thread
    size_t w = (size_t)blockIdx.x * blockDim.x + threadIdx.x;   // word index
    const unsigned char* p = mask + w * 32;
    unsigned bits = 0;
    #pragma unroll
    for (int j = 0; j < 8; ++j) {
        unsigned v4 = *(const unsigned*)(p + 4 * j);  // 4 bytes
        #pragma unroll
        for (int c = 0; c < 4; ++c)
            bits |= (((v4 >> (8 * c)) & 0xffu) ? 1u : 0u) << (4 * j + c);
    }
    mbits[w] = bits;
}

// ---------------------------------------------------------------------------
// Kernel 1: feature maps.  One block = 16 query/key rows of one (b,h).
// Produces |qf| and |kf| as bf16 [B,H,S,DKER] in workspace.
// ---------------------------------------------------------------------------
__global__ __launch_bounds__(256) void feat_kernel(
    const float* __restrict__ q, const float* __restrict__ k,
    const float* __restrict__ Wq1, const float* __restrict__ Wk1,
    const float* __restrict__ Wq2, const float* __restrict__ Wk2,
    const float* __restrict__ Wint,
    const float* __restrict__ scalingD, const float* __restrict__ scalingD2,
    bf16* __restrict__ absqf, bf16* __restrict__ abskf)
{
    __shared__ float qrow[16][64];
    __shared__ float krow[16][64];
    __shared__ float hq[16][256];
    __shared__ float hk[16][256];
    __shared__ float kf2s[16][64];

    const int t  = threadIdx.x;
    const int s0 = blockIdx.x * 16;
    const int h  = blockIdx.y;
    const int b  = blockIdx.z;

    #pragma unroll
    for (int i = 0; i < 4; ++i) {
        int idx = t + 256 * i;
        int r = idx >> 6, c = idx & 63;
        size_t g = ((size_t)(b * Sc + s0 + r)) * Dc + h * DHc + c;
        qrow[r][c] = q[g];
        krow[r][c] = k[g];
    }
    __syncthreads();

    {   // stage 1: hidden unit e = t (256 units), all 16 rows
        float aq[16], ak[16];
        #pragma unroll
        for (int r = 0; r < 16; ++r) { aq[r] = 0.f; ak[r] = 0.f; }
        const float* wq1p = Wq1 + ((size_t)h * DHc) * DHIDc + t;
        const float* wk1p = Wk1 + ((size_t)h * DHc) * DHIDc + t;
        for (int d = 0; d < DHc; ++d) {
            float wq = wq1p[d * DHIDc];
            float wk = wk1p[d * DHIDc];
            #pragma unroll
            for (int r = 0; r < 16; ++r) {
                aq[r] += qrow[r][d] * wq;
                ak[r] += krow[r][d] * wk;
            }
        }
        #pragma unroll
        for (int r = 0; r < 16; ++r) {
            hq[r][t] = gelu_tanh(aq[r]);
            hk[r][t] = gelu_tanh(ak[r]);
        }
    }
    __syncthreads();

    const int o = t & 63, g = t >> 6;
    {   // stage 2: output unit o, rows g, g+4, g+8, g+12
        float bq[4] = {0.f, 0.f, 0.f, 0.f};
        float bk[4] = {0.f, 0.f, 0.f, 0.f};
        const float* wq2p = Wq2 + ((size_t)h * DHIDc) * DKERc + o;
        const float* wk2p = Wk2 + ((size_t)h * DHIDc) * DKERc + o;
        for (int e = 0; e < DHIDc; ++e) {
            float w2q = wq2p[e * DKERc];
            float w2k = wk2p[e * DKERc];
            #pragma unroll
            for (int j = 0; j < 4; ++j) {
                int r = g + 4 * j;
                bq[j] += hq[r][e] * w2q;
                bk[j] += hk[r][e] * w2k;
            }
        }
        float sD = fabsf(scalingD[h * DKERc + o]);
        #pragma unroll
        for (int j = 0; j < 4; ++j) {
            int r = g + 4 * j;
            float qf = gelu_tanh(bq[j]);
            absqf[((size_t)(b * Hc + h) * Sc + s0 + r) * DKERc + o] = (bf16)fabsf(qf);
            kf2s[r][o] = sD * gelu_tanh(bk[j]);
        }
    }
    __syncthreads();

    {   // stage 3: kf += (kf @ Wint) * scalingD2 ; store |kf|
        float ck[4] = {0.f, 0.f, 0.f, 0.f};
        const float* wip = Wint + ((size_t)h * DKERc) * DKERc + o;
        for (int e = 0; e < DKERc; ++e) {
            float wi = wip[e * DKERc];
            #pragma unroll
            for (int j = 0; j < 4; ++j) ck[j] += kf2s[g + 4 * j][e] * wi;
        }
        float sD2 = scalingD2[h * DKERc + o];
        #pragma unroll
        for (int j = 0; j < 4; ++j) {
            int r = g + 4 * j;
            float kf = kf2s[r][o] + ck[j] * sD2;
            abskf[((size_t)(b * Hc + h) * Sc + s0 + r) * DKERc + o] = (bf16)fabsf(kf);
        }
    }
}

// ---------------------------------------------------------------------------
// Kernel 2: repack v into bf16 vhT[B,H,DH,S]; PV B-fragments become
// contiguous 16B chunks.  LDS tile transpose, coalesced both sides.
// ---------------------------------------------------------------------------
__global__ __launch_bounds__(256) void vtrans_kernel(
    const float* __restrict__ v, bf16* __restrict__ vhT)
{
    __shared__ float tile[64][65];
    const int t  = threadIdx.x;
    const int s0 = blockIdx.x * 64;
    const int h  = blockIdx.y;
    const int b  = blockIdx.z;

    #pragma unroll
    for (int i = 0; i < 16; ++i) {
        int idx = t + 256 * i;
        int sl = idx >> 6, c = idx & 63;
        tile[sl][c] = v[((size_t)(b * Sc + s0 + sl)) * Dc + h * DHc + c];
    }
    __syncthreads();
    size_t base = ((size_t)(b * Hc + h)) * DHc;
    #pragma unroll
    for (int i = 0; i < 16; ++i) {
        int idx = t + 256 * i;
        int dh = idx >> 6, sl = idx & 63;
        vhT[(base + dh) * Sc + s0 + sl] = (bf16)tile[sl][dh];
    }
}

// ---------------------------------------------------------------------------
// Kernel 3: fused attention.  128 threads = 4 waves; wave wv owns 16 query
// rows.  Pass 1: WMMA scores + masked row sums -> logaddexp norm.
// Pass 2: recompute scores, w = mask ? (s+eps)e^-norm : e^(saw-norm),
// P -> bf16 via wave-private LDS transpose (no barriers needed: LDS ops of a
// wave are in-order), PV via WMMA.  The S x S w matrix never touches HBM.
// ---------------------------------------------------------------------------
__global__ __launch_bounds__(128) void attn_kernel(
    const bf16* __restrict__ absqf, const bf16* __restrict__ abskf,
    const bf16* __restrict__ vhT,
    const unsigned* __restrict__ mbits,         // [B,S,S/32] packed mask
    const float* __restrict__ sparse_lse,       // [B,H,S,1]
    const float* __restrict__ saw,              // [B,H,S,S]
    float* __restrict__ out)                    // [B,S,D]
{
    // 80B rows: 16B-aligned for ds_load_b128, stride 20 dwords -> no conflicts
    __shared__ bf16 wtile[4][16][40];

    const int lane = threadIdx.x & 31;
    const int wv   = threadIdx.x >> 5;
    const int b    = blockIdx.z;
    const int h    = blockIdx.y;
    const int bh   = b * Hc + h;
    const int q0   = blockIdx.x * 64 + wv * 16;
    const int nl   = lane & 15;
    const int koff = (lane & 16) ? 8 : 0;   // A/B lane-half K offset
    const int moff = (lane & 16) ? 8 : 0;   // C/D lane-half M offset

    // A fragments: |qf| rows, K = 0..31 and 32..63
    const bf16* qrowp = absqf + ((size_t)bh * Sc + q0 + nl) * DKERc;
    v16bf aq0 = load_frag(qrowp, koff);
    v16bf aq1 = load_frag(qrowp + 32, koff);

    const bf16* kfb = abskf + (size_t)bh * Sc * DKERc;
    // this lane carries the packed mask word of query row q0+nl
    const unsigned* mrowp = mbits + ((size_t)b * Sc + q0 + nl) * (Sc / 32);

    // ---- pass 1: masked row sums over all keys ----
    float acc[8];
    #pragma unroll
    for (int i = 0; i < 8; ++i) acc[i] = 0.f;

    for (int kt = 0; kt < Sc / 32; ++kt) {
        int kb = kt * 32;
        const bf16* kr0 = kfb + (size_t)(kb + nl) * DKERc;
        const bf16* kr1 = kfb + (size_t)(kb + 16 + nl) * DKERc;
        v8f d0 = {}, d1 = {};
        d0 = wmma_bf16(aq0, load_frag(kr0, koff), d0);
        d0 = wmma_bf16(aq1, load_frag(kr0 + 32, koff), d0);
        d1 = wmma_bf16(aq0, load_frag(kr1, koff), d1);
        d1 = wmma_bf16(aq1, load_frag(kr1 + 32, koff), d1);

        unsigned wbits = mrowp[kt];   // bits for keys kb..kb+31 of row q0+nl
        #pragma unroll
        for (int i = 0; i < 8; ++i) {
            unsigned rb = __shfl(wbits, i + moff, 32);  // row q0+i+moff bits
            acc[i] += ((rb >> nl) & 1u)        ? d0[i] : 0.0f;
            acc[i] += ((rb >> (16 + nl)) & 1u) ? d1[i] : 0.0f;
        }
    }

    // reduce across the 16-lane half, build per-row norm / exp(-norm)
    float norm[8], inv[8];
    #pragma unroll
    for (int i = 0; i < 8; ++i) {
        float s = acc[i];
        s += __shfl_xor(s, 1, 32);
        s += __shfl_xor(s, 2, 32);
        s += __shfl_xor(s, 4, 32);
        s += __shfl_xor(s, 8, 32);
        float lr = __logf(s + 1e-6f);
        float sp = sparse_lse[(size_t)bh * Sc + q0 + i + moff];
        float mx = fmaxf(lr, sp);
        float nm = mx + log1pf(__expf(-fabsf(lr - sp)));
        norm[i] = nm;
        inv[i]  = __expf(-nm);
    }

    // ---- pass 2: recompute scores, form P, accumulate P @ V ----
    const float* sawb = saw + (size_t)bh * Sc * Sc;
    v8f oacc[4] = {};

    for (int kt = 0; kt < Sc / 32; ++kt) {
        int kb = kt * 32;
        // prefetch the sparse-weight stream two tiles ahead (one 128B line
        // per query row; lanes 0-15 cover the wave's 16 rows)
        __builtin_prefetch(sawb + (size_t)(q0 + nl) * Sc + kb + 64, 0, 0);

        const bf16* kr0 = kfb + (size_t)(kb + nl) * DKERc;
        const bf16* kr1 = kfb + (size_t)(kb + 16 + nl) * DKERc;
        v8f d0 = {}, d1 = {};
        d0 = wmma_bf16(aq0, load_frag(kr0, koff), d0);
        d0 = wmma_bf16(aq1, load_frag(kr0 + 32, koff), d0);
        d1 = wmma_bf16(aq0, load_frag(kr1, koff), d1);
        d1 = wmma_bf16(aq1, load_frag(kr1 + 32, koff), d1);

        unsigned wbits = mrowp[kt];
        #pragma unroll
        for (int i = 0; i < 8; ++i) {
            int m = i + moff;
            unsigned rb = __shfl(wbits, i + moff, 32);
            size_t roff = (size_t)(q0 + m) * Sc;
            size_t j0 = roff + kb + nl;
            size_t j1 = roff + kb + 16 + nl;
            float w0 = ((rb >> nl) & 1u)
                         ? (d0[i] + 1e-6f) * inv[i]
                         : __expf(sawb[j0] - norm[i]);
            float w1 = ((rb >> (16 + nl)) & 1u)
                         ? (d1[i] + 1e-6f) * inv[i]
                         : __expf(sawb[j1] - norm[i]);
            wtile[wv][m][nl]      = (bf16)w0;
            wtile[wv][m][16 + nl] = (bf16)w1;
        }
        // wave-private LDS region + in-order LDS pipe: no barrier required

        v16bf aw;   // read P back as an A fragment (row = nl, 32-wide K)
        {
            union { v16bf v; v8bf hh[2]; } u;
            u.hh[0] = *(const v8bf*)(&wtile[wv][nl][koff]);
            u.hh[1] = *(const v8bf*)(&wtile[wv][nl][16 + koff]);
            aw = u.v;
        }

        #pragma unroll
        for (int nb = 0; nb < 4; ++nb) {
            const bf16* vrow = vhT + ((size_t)bh * DHc + nb * 16 + nl) * Sc + kb;
            v16bf bv = load_frag(vrow, koff);
            oacc[nb] = wmma_bf16(aw, bv, oacc[nb]);
        }
    }

    // store O: [16 rows x 64 dh] -> out[b, s, h*64 + dh]
    #pragma unroll
    for (int nb = 0; nb < 4; ++nb) {
        #pragma unroll
        for (int i = 0; i < 8; ++i) {
            int m = q0 + i + moff;
            out[((size_t)(b * Sc + m)) * Dc + h * DHc + nb * 16 + nl] = oacc[nb][i];
        }
    }
}

// ---------------------------------------------------------------------------
extern "C" void kernel_launch(void* const* d_in, const int* in_sizes, int n_in,
                              void* d_out, int out_size, void* d_ws, size_t ws_size,
                              hipStream_t stream) {
    (void)in_sizes; (void)n_in; (void)out_size; (void)ws_size;
    // input order per setup_inputs(): x_t, q, k, v, lr_attn_mask,
    // sparse_norms_lse, sparse_attn_weights, lambda_constant,
    // Wq1, Wk1, Wq2, Wk2, Wint, scalingD, scalingD2
    const float* q    = (const float*)d_in[1];
    const float* k    = (const float*)d_in[2];
    const float* v    = (const float*)d_in[3];
    const unsigned char* mask = (const unsigned char*)d_in[4];   // jax bool = 1B
    const float* slse = (const float*)d_in[5];
    const float* saw  = (const float*)d_in[6];
    const float* Wq1  = (const float*)d_in[8];
    const float* Wk1  = (const float*)d_in[9];
    const float* Wq2  = (const float*)d_in[10];
    const float* Wk2  = (const float*)d_in[11];
    const float* Wint = (const float*)d_in[12];
    const float* sD   = (const float*)d_in[13];
    const float* sD2  = (const float*)d_in[14];
    float* out = (float*)d_out;

    // workspace: |qf| (8MB) | |kf| (8MB) | vhT (8MB) | packed mask (1MB)
    const size_t feat_elems = (size_t)Bc * Hc * Sc * DKERc;
    bf16* absqf = (bf16*)d_ws;
    bf16* abskf = absqf + feat_elems;
    bf16* vhT   = abskf + feat_elems;
    unsigned* mbits = (unsigned*)(vhT + feat_elems);

    const int nwords = Bc * Sc * (Sc / 32);   // 262144
    maskpack_kernel<<<nwords / 256, 256, 0, stream>>>(mask, mbits);
    feat_kernel<<<dim3(Sc / 16, Hc, Bc), 256, 0, stream>>>(
        q, k, Wq1, Wk1, Wq2, Wk2, Wint, sD, sD2, absqf, abskf);
    vtrans_kernel<<<dim3(Sc / 64, Hc, Bc), 256, 0, stream>>>(v, vhT);
    attn_kernel<<<dim3(Sc / 64, Hc, Bc), 128, 0, stream>>>(
        absqf, abskf, vhT, mbits, slse, saw, out);
}
